// SelfAttention_8418135900720
// MI455X (gfx1250) — matrix operand order repaired and checked by reading, not compiled, not run on hardware
//
#include <hip/hip_runtime.h>
#include <hip/hip_bf16.h>
#include <math.h>

typedef _Float16 half_t;
typedef __attribute__((ext_vector_type(16))) _Float16 v16h;
typedef __attribute__((ext_vector_type(8)))  float    v8f;

#define BATCH 8
#define DIM 512
#define HEADS 8
#define DHEAD 64
#define NPIX 1024              // 32*32 pixels per batch
#define NROWS (BATCH*NPIX)     // 8192 total pixel rows
#define NKPAD 1056             // 1 null + 1024 keys + 31 pad = 33 * 32
#define NJT (NKPAD/32)         // 33 key tiles of 32
#define ATT_SCALE 0.125f       // 64^-0.5

// ---- WMMA fragment index maps as used here (A/B in contiguous k order, wave32) ----
// A (16x32 f16): lane l holds row m=l&15; half t (0..15) maps to
//   k = t + 16*(l>>4)
__device__ __forceinline__ int a_kidx(int t, int hi) {
  return t + 16 * hi;
}
// B (32x16 f16): lane l holds col n=l&15; half t maps to k = t + 16*(l>>4)
// C/D (16x16 f32): lane l holds col n=l&15; vgpr j holds row m = 8*(l>>4)+j

__device__ __forceinline__ v8f wmma16(v16h a, v16h b, v8f c) {
  return __builtin_amdgcn_wmma_f32_16x16x32_f16(false, a, false, b, (short)0, c,
                                                false, false);
}
__device__ __forceinline__ v8f zero8() {
  v8f z = {0.f, 0.f, 0.f, 0.f, 0.f, 0.f, 0.f, 0.f};
  return z;
}

// CDNA5 async copy global->LDS (ASYNCcnt-tracked). VDST carries the LDS byte
// offset (low 32 bits of a generic pointer into __shared__), VADDR the 64-bit
// global address.  (CDNA5 ISA)
__device__ __forceinline__ void async_copy_b128(const half_t* gsrc,
                                                half_t* ldst) {
  unsigned lds_off = (unsigned)(size_t)ldst;
  asm volatile("global_load_async_to_lds_b128 %0, %1, off" ::"v"(lds_off),
               "v"(gsrc)
               : "memory");
}

// ---------------------------------------------------------------------------
// Kernel 1: convert f32 weights to f16 once.
// ---------------------------------------------------------------------------
__global__ void convert_weights_kernel(const float* __restrict__ wq,
                                       const float* __restrict__ wv,
                                       const float* __restrict__ wo,
                                       half_t* __restrict__ wq16,
                                       half_t* __restrict__ wv16,
                                       half_t* __restrict__ wo16) {
  int i = blockIdx.x * blockDim.x + threadIdx.x;
  if (i < DIM * DIM) {
    wq16[i] = (half_t)wq[i];
    wv16[i] = (half_t)wv[i];
    wo16[i] = (half_t)wo[i];
  }
}

// ---------------------------------------------------------------------------
// Kernel 2: channel RMSNorm; one wave per pixel. Output f16 Xn[8192][512].
// ---------------------------------------------------------------------------
__global__ void __launch_bounds__(256) rmsnorm_kernel(
    const float* __restrict__ fmap, const float* __restrict__ gamma,
    half_t* __restrict__ Xn) {
  int wave = threadIdx.x >> 5;
  int lane = threadIdx.x & 31;
  int g = blockIdx.x * 8 + wave;  // global pixel row 0..8191
  int b = g >> 10;
  int p = g & 1023;
  const float* src = fmap + (size_t)b * DIM * NPIX + p;
  float vals[16];
  float ss = 0.f;
#pragma unroll
  for (int i = 0; i < 16; ++i) {
    int c = lane + i * 32;
    float v = src[(size_t)c * NPIX];
    vals[i] = v;
    ss += v * v;
  }
#pragma unroll
  for (int off = 16; off >= 1; off >>= 1) ss += __shfl_xor(ss, off, 32);
  // x = fmap / max(||fmap||_2, 1e-12) * sqrt(512) * gamma
  float inv = 22.62741699796952f / fmaxf(sqrtf(ss), 1e-12f);
  half_t* dst = Xn + (size_t)g * DIM;
#pragma unroll
  for (int i = 0; i < 16; ++i) {
    int c = lane + i * 32;
    dst[c] = (half_t)(vals[i] * inv * gamma[c]);
  }
}

// ---------------------------------------------------------------------------
// Kernel 3: Q = Xn*Wq^T, V = Xn*Wv^T (shared A frags). Q goes into Kpad rows
// 1..1024 of its (b,h) slab (k = q); V into Vpad rows 1..1024.
// Wave task = 16-row x 64-col output tile; K loop 512 in steps of 32.
// ---------------------------------------------------------------------------
__global__ void __launch_bounds__(256) proj_gemm_kernel(
    const half_t* __restrict__ Xn, const half_t* __restrict__ Wq16,
    const half_t* __restrict__ Wv16, half_t* __restrict__ Kpad,
    half_t* __restrict__ Vpad) {
  int wave = threadIdx.x >> 5;
  int lane = threadIdx.x & 31;
  int hi = lane >> 4;
  int ln = lane & 15;
  int task = blockIdx.x * 8 + wave;  // 4096 tasks
  int rowTile = task >> 3;           // 0..511
  int cg = task & 7;                 // 0..7 (64-wide col group)
  int r0 = rowTile * 16;
  int o0 = cg * 64;

  v8f accQ[4], accV[4];
#pragma unroll
  for (int s = 0; s < 4; ++s) {
    accQ[s] = zero8();
    accV[s] = zero8();
  }

  for (int kb = 0; kb < DIM; kb += 32) {
    v16h a;
    const half_t* arow = Xn + (size_t)(r0 + ln) * DIM + kb;
#pragma unroll
    for (int t = 0; t < 16; ++t) a[t] = arow[a_kidx(t, hi)];
#pragma unroll
    for (int s = 0; s < 4; ++s) {
      int o = o0 + s * 16 + ln;
      const half_t* qrow = Wq16 + (size_t)o * DIM + kb + 16 * hi;
      const half_t* vrow = Wv16 + (size_t)o * DIM + kb + 16 * hi;
      v16h bq, bv;
#pragma unroll
      for (int t = 0; t < 16; ++t) {
        bq[t] = qrow[t];
        bv[t] = vrow[t];
      }
      accQ[s] = wmma16(a, bq, accQ[s]);
      accV[s] = wmma16(a, bv, accV[s]);
    }
  }

  int b = r0 >> 10;  // tile never crosses a batch boundary (1024 % 16 == 0)
#pragma unroll
  for (int s = 0; s < 4; ++s) {
    int o = o0 + s * 16 + ln;
    int h = o >> 6, d = o & 63;
#pragma unroll
    for (int j = 0; j < 8; ++j) {
      int r = r0 + 8 * hi + j;
      int pl = r & 1023;
      size_t idx = ((size_t)(b * HEADS + h) * NKPAD + 1 + pl) * DHEAD + d;
      Kpad[idx] = (half_t)accQ[s][j];
      Vpad[idx] = (half_t)accV[s][j];
    }
  }
}

// ---------------------------------------------------------------------------
// Kernel 4: fill null token (row 0), zero pad rows (1025..1055), compute |k|^2
// per row (1e30 for pad rows -> softmax weight exactly 0).
// ---------------------------------------------------------------------------
__global__ void prep_kernel(const float* __restrict__ null_kv,
                            half_t* __restrict__ Kpad,
                            half_t* __restrict__ Vpad,
                            float* __restrict__ ksq) {
  int rid = blockIdx.x * blockDim.x + threadIdx.x;
  if (rid >= BATCH * HEADS * NKPAD) return;
  int bh = rid / NKPAD;
  int j = rid - bh * NKPAD;
  int h = bh & 7;
  half_t* krow = Kpad + (size_t)rid * DHEAD;
  half_t* vrow = Vpad + (size_t)rid * DHEAD;
  if (j == 0) {
    float s = 0.f;
    for (int d = 0; d < DHEAD; ++d) {
      float kv = null_kv[h * DHEAD + d];
      float vv = null_kv[HEADS * DHEAD + h * DHEAD + d];
      krow[d] = (half_t)kv;
      vrow[d] = (half_t)vv;
      s += kv * kv;
    }
    ksq[rid] = s;
  } else if (j <= NPIX) {
    float s = 0.f;
    for (int d = 0; d < DHEAD; ++d) {
      float kv = (float)krow[d];
      s += kv * kv;
    }
    ksq[rid] = s;
  } else {
    for (int d = 0; d < DHEAD; ++d) {
      krow[d] = (half_t)0.f;
      vrow[d] = (half_t)0.f;
    }
    ksq[rid] = 1e30f;  // sim -> -1.25e29 -> exp underflows to 0
  }
}

// ---------------------------------------------------------------------------
// Kernel 5: attention. Grid = 64 (b,h) x 8 query-chunks = 512 blocks; each of
// the 8 waves owns one 16-query tile. The 32-key K/V tiles (4 KB each) are
// staged into LDS cooperatively with global_load_async_to_lds_b128 and
// double-buffered: stage(jt+1) is issued, then s_wait_asynccnt 2 drains
// stage(jt) while the next copy is in flight. All 8 waves share each staged
// tile (8x traffic reduction vs per-wave global loads).
// Online softmax kept in f32 in the C-layout register mapping; P transposed
// C-layout -> A-layout through a per-wave LDS staging tile.
// ---------------------------------------------------------------------------
__global__ void __launch_bounds__(256) attn_kernel(
    const half_t* __restrict__ Kpad, const half_t* __restrict__ Vpad,
    const float* __restrict__ ksq, half_t* __restrict__ O16) {
  __shared__ __align__(16) half_t kbuf[2][32 * DHEAD];
  __shared__ __align__(16) half_t vbuf[2][32 * DHEAD];
  __shared__ __align__(16) half_t pstage[8][16 * 32];

  int tid = threadIdx.x;
  int wave = tid >> 5;
  int lane = tid & 31;
  int hi = lane >> 4;
  int ln = lane & 15;
  int bh = blockIdx.x >> 3;  // 0..63
  int qc = blockIdx.x & 7;   // query chunk
  int b = bh >> 3, h = bh & 7;
  const half_t* K = Kpad + (size_t)bh * NKPAD * DHEAD;
  const half_t* V = Vpad + (size_t)bh * NKPAD * DHEAD;
  const float* Ks = ksq + (size_t)bh * NKPAD;
  half_t* pst = &pstage[wave][0];

  // cooperative stage helper mapping: thread -> (row = tid>>3, 16B chunk)
  int srow = tid >> 3;
  int scol = (tid & 7) * 8;

  int qt = qc * 8 + wave;  // this wave's query tile, 0..63
  int i0 = qt * 16;

  // Q A-fragments (d 0..31 and 32..63); query i lives at Kpad row 1+i
  v16h aQ0, aQ1;
  const half_t* qrow = K + (size_t)(1 + i0 + ln) * DHEAD;
#pragma unroll
  for (int t = 0; t < 16; ++t) {
    aQ0[t] = qrow[a_kidx(t, hi)];
    aQ1[t] = qrow[32 + a_kidx(t, hi)];
  }
  float qs[8];
#pragma unroll
  for (int j = 0; j < 8; ++j) qs[j] = Ks[1 + i0 + 8 * hi + j];

  float mrow[8], lrow[8];
#pragma unroll
  for (int j = 0; j < 8; ++j) {
    mrow[j] = -3.0e38f;
    lrow[j] = 0.f;
  }
  v8f accO[4];
#pragma unroll
  for (int s = 0; s < 4; ++s) accO[s] = zero8();

  // prologue: stage tile 0 into buffer 0 (2 async ops per wave)
  async_copy_b128(K + (size_t)srow * DHEAD + scol, &kbuf[0][srow * DHEAD + scol]);
  async_copy_b128(V + (size_t)srow * DHEAD + scol, &vbuf[0][srow * DHEAD + scol]);

  for (int jt = 0; jt < NJT; ++jt) {
    int cur = jt & 1;
    int j0 = jt * 32;
    if (jt + 1 < NJT) {
      size_t g = (size_t)((jt + 1) * 32 + srow) * DHEAD + scol;
      async_copy_b128(K + g, &kbuf[1 - cur][srow * DHEAD + scol]);
      async_copy_b128(V + g, &vbuf[1 - cur][srow * DHEAD + scol]);
      asm volatile("s_wait_asynccnt 0x2" ::: "memory");  // drain current tile
    } else {
      asm volatile("s_wait_asynccnt 0x0" ::: "memory");
    }
    __syncthreads();  // current tile visible to all waves

    // --- S = Q @ K^T for 32 keys (two 16-wide subtiles), K frags from LDS ---
    v8f S[2];
    float kcol[2];
#pragma unroll
    for (int u = 0; u < 2; ++u) {
      const half_t* krow = &kbuf[cur][(u * 16 + ln) * DHEAD + 16 * hi];
      v16h b0, b1;
#pragma unroll
      for (int t = 0; t < 16; ++t) {
        b0[t] = krow[t];       // d = 16*hi + t      (pairs with aQ0)
        b1[t] = krow[32 + t];  // d = 32 + 16*hi + t (pairs with aQ1)
      }
      v8f sacc = zero8();
      sacc = wmma16(aQ0, b0, sacc);
      sacc = wmma16(aQ1, b1, sacc);
      S[u] = sacc;
      kcol[u] = Ks[j0 + u * 16 + ln];
    }
    // sim = scale * (2*qk - |q|^2 - |k|^2); online softmax row stats
    float sim[2][8], rmax[8];
#pragma unroll
    for (int j = 0; j < 8; ++j) {
      float s0 = ATT_SCALE * (2.f * S[0][j] - qs[j] - kcol[0]);
      float s1 = ATT_SCALE * (2.f * S[1][j] - qs[j] - kcol[1]);
      sim[0][j] = s0;
      sim[1][j] = s1;
      rmax[j] = fmaxf(s0, s1);
    }
#pragma unroll
    for (int off = 8; off >= 1; off >>= 1)
#pragma unroll
      for (int j = 0; j < 8; ++j)
        rmax[j] = fmaxf(rmax[j], __shfl_xor(rmax[j], off, 32));
    float alpha[8], rsum[8];
#pragma unroll
    for (int j = 0; j < 8; ++j) {
      float mn = fmaxf(mrow[j], rmax[j]);
      alpha[j] = __expf(mrow[j] - mn);
      mrow[j] = mn;
      float p0 = __expf(sim[0][j] - mn);
      float p1 = __expf(sim[1][j] - mn);
      sim[0][j] = p0;
      sim[1][j] = p1;
      rsum[j] = p0 + p1;
    }
#pragma unroll
    for (int off = 8; off >= 1; off >>= 1)
#pragma unroll
      for (int j = 0; j < 8; ++j) rsum[j] += __shfl_xor(rsum[j], off, 32);
#pragma unroll
    for (int j = 0; j < 8; ++j) lrow[j] = lrow[j] * alpha[j] + rsum[j];
#pragma unroll
    for (int s = 0; s < 4; ++s)
#pragma unroll
      for (int j = 0; j < 8; ++j) accO[s][j] *= alpha[j];

    // Stage P (f32 C-layout) -> LDS -> f16 A-layout fragment (per-wave tile)
#pragma unroll
    for (int u = 0; u < 2; ++u)
#pragma unroll
      for (int j = 0; j < 8; ++j)
        pst[(8 * hi + j) * 32 + u * 16 + ln] = (half_t)sim[u][j];
    asm volatile("s_wait_dscnt 0x0" ::: "memory");
    v16h aP;
#pragma unroll
    for (int t = 0; t < 16; ++t) aP[t] = pst[ln * 32 + a_kidx(t, hi)];

    // O += P @ V (K=32 keys, N=64 head dims), V frags from LDS
#pragma unroll
    for (int s = 0; s < 4; ++s) {
      const half_t* vcol = &vbuf[cur][s * 16 + ln];
      v16h bV;
#pragma unroll
      for (int t = 0; t < 16; ++t) bV[t] = vcol[(t + 16 * hi) * DHEAD];
      accO[s] = wmma16(aP, bV, accO[s]);
    }
    __syncthreads();  // all waves done with `cur` before it is re-staged
  }

  // epilogue: normalize and store O16[pixel_row][h*64 + d]
#pragma unroll
  for (int s = 0; s < 4; ++s)
#pragma unroll
    for (int j = 0; j < 8; ++j) {
      int m = 8 * hi + j;
      int r = b * NPIX + i0 + m;
      int o = h * DHEAD + s * 16 + ln;
      O16[(size_t)r * DIM + o] = (half_t)(accO[s][j] / lrow[j]);
    }
}

// ---------------------------------------------------------------------------
// Kernel 6: final[b][c][p] = sum_o wo[c][o] * O[p][o], f32 out.
// ---------------------------------------------------------------------------
__global__ void __launch_bounds__(256) out_gemm_kernel(
    const half_t* __restrict__ Wo16, const half_t* __restrict__ O16,
    float* __restrict__ out) {
  int wave = threadIdx.x >> 5;
  int lane = threadIdx.x & 31;
  int hi = lane >> 4, ln = lane & 15;
  int task = blockIdx.x * 8 + wave;  // 4096 tasks
  int bb = task >> 9;
  int rem = task & 511;
  int ct = rem >> 4;  // 0..31: 16-row tile of output channels
  int pg = rem & 15;  // 0..15: 64-wide pixel col group
  int c0 = ct * 16;
  int p0 = pg * 64;
  v8f acc[4];
#pragma unroll
  for (int s = 0; s < 4; ++s) acc[s] = zero8();

  for (int kb = 0; kb < DIM; kb += 32) {
    v16h a;
    const half_t* arow = Wo16 + (size_t)(c0 + ln) * DIM + kb;
#pragma unroll
    for (int t = 0; t < 16; ++t) a[t] = arow[a_kidx(t, hi)];
#pragma unroll
    for (int s = 0; s < 4; ++s) {
      const half_t* brow =
          O16 + (size_t)(bb * NPIX + p0 + s * 16 + ln) * DIM + kb + 16 * hi;
      v16h bf;
#pragma unroll
      for (int t = 0; t < 16; ++t) bf[t] = brow[t];
      acc[s] = wmma16(a, bf, acc[s]);
    }
  }
#pragma unroll
  for (int s = 0; s < 4; ++s)
#pragma unroll
    for (int j = 0; j < 8; ++j) {
      int c = c0 + 8 * hi + j;
      int p = p0 + s * 16 + ln;
      out[((size_t)bb * DIM + c) * NPIX + p] = acc[s][j];
    }
}

// ---------------------------------------------------------------------------
// Workspace layout (bytes):
//  Xn    @ 0         : 8192*512*2   = 8,388,608
//  Wq16  @ 8388608   : 512*512*2    =   524,288
//  Wv16  @ 8912896   : 512*512*2    =   524,288
//  Wo16  @ 9437184   : 512*512*2    =   524,288
//  Kpad  @ 9961472   : 64*1056*64*2 = 8,650,752
//  Vpad  @ 18612224  : 64*1056*64*2 = 8,650,752
//  ksq   @ 27262976  : 64*1056*4    =   270,336
//  O16   @ 27533312  : 8192*512*2   = 8,388,608   (end = 35,921,920)
// ---------------------------------------------------------------------------
extern "C" void kernel_launch(void* const* d_in, const int* in_sizes, int n_in,
                              void* d_out, int out_size, void* d_ws,
                              size_t ws_size, hipStream_t stream) {
  const float* fmap = (const float*)d_in[0];
  const float* gamma = (const float*)d_in[1];
  const float* wq = (const float*)d_in[2];
  const float* wv = (const float*)d_in[3];
  const float* nullkv = (const float*)d_in[4];
  const float* wo = (const float*)d_in[5];
  float* out = (float*)d_out;

  char* ws = (char*)d_ws;
  half_t* Xn = (half_t*)(ws + 0);
  half_t* Wq16 = (half_t*)(ws + 8388608);
  half_t* Wv16 = (half_t*)(ws + 8912896);
  half_t* Wo16 = (half_t*)(ws + 9437184);
  half_t* Kpad = (half_t*)(ws + 9961472);
  half_t* Vpad = (half_t*)(ws + 18612224);
  float* ksq = (float*)(ws + 27262976);
  half_t* O16 = (half_t*)(ws + 27533312);

  convert_weights_kernel<<<(DIM * DIM + 255) / 256, 256, 0, stream>>>(
      wq, wv, wo, Wq16, Wv16, Wo16);
  rmsnorm_kernel<<<NROWS / 8, 256, 0, stream>>>(fmap, gamma, Xn);
  proj_gemm_kernel<<<512, 256, 0, stream>>>(Xn, Wq16, Wv16, Kpad, Vpad);
  prep_kernel<<<(BATCH * HEADS * NKPAD + 255) / 256, 256, 0, stream>>>(
      nullkv, Kpad, Vpad, ksq);
  attn_kernel<<<BATCH * HEADS * 8, 256, 0, stream>>>(Kpad, Vpad, ksq, O16);
  out_gemm_kernel<<<512, 256, 0, stream>>>(Wo16, O16, out);
}
